// DySPN_32384053412611
// MI455X (gfx1250) — matrix operand order repaired and checked by reading, not compiled, not run on hardware
//
#include <hip/hip_runtime.h>
#include <cstdint>

#define H_ 352
#define W_ 1216
#define C_ 64
#define B_ 2

typedef __attribute__((ext_vector_type(16))) __bf16 v16bf;
typedef __attribute__((ext_vector_type(8)))  float  v8f;
typedef unsigned int u32x4 __attribute__((ext_vector_type(4)));

#if defined(__has_builtin)
#if __has_builtin(__builtin_amdgcn_global_load_async_to_lds_b128)
#define HAVE_ASYNC_LDS 1
#endif
#endif

#ifdef HAVE_ASYNC_LDS
typedef int v4i __attribute__((vector_size(16)));
typedef __attribute__((address_space(1))) v4i* gptr_v4i;
typedef __attribute__((address_space(3))) v4i* lptr_v4i;
__device__ __forceinline__ void async_copy_b128(const void* g, void* l) {
  __builtin_amdgcn_global_load_async_to_lds_b128(
      (gptr_v4i)g, (lptr_v4i)l, 0, 0);
}
#endif

__device__ __forceinline__ unsigned short f2bf(float f) {
  unsigned int u = __builtin_bit_cast(unsigned int, f);
  u += 0x7fffu + ((u >> 16) & 1u);   // round-to-nearest-even
  return (unsigned short)(u >> 16);
}

// ---------------------------------------------------------------------------
// Kernel 0: fold BN into conv weights, pack bf16 B-matrix [n][k], k = tap*64+c
// n: 0..23 aff7 | 24..39 aff5 | 40..47 aff3 | 48..51 att7/5/3/1 ch5 | 52..63 zero
// ---------------------------------------------------------------------------
struct PackArgs {
  const float* w[7]; const float* g[7]; const float* b[7];
  const float* m[7]; const float* v[7];
};

__global__ void pack_weights(PackArgs pa, unsigned short* __restrict__ wpack,
                             float* __restrict__ bias) {
  const int n = threadIdx.x;
  if (n >= 64) return;
  int set = -1, oc = 0;
  if (n < 24)      { set = 0; oc = n; }
  else if (n < 40) { set = 1; oc = n - 24; }
  else if (n < 48) { set = 2; oc = n - 40; }
  else if (n < 52) { set = 3 + (n - 48); oc = 5; }   // att channel it=5 (only live iter)
  if (set < 0) {
    bias[n] = 0.f;
    for (int k = 0; k < 576; ++k) wpack[n * 576 + k] = 0;
    return;
  }
  const float s = pa.g[set][oc] / sqrtf(pa.v[set][oc] + 1e-5f);
  bias[n] = pa.b[set][oc] - pa.m[set][oc] * s;
  const float* w = pa.w[set] + (size_t)oc * C_ * 9;     // OIHW: w[oc][c][ki][kj]
  for (int tap = 0; tap < 9; ++tap)
    for (int c = 0; c < C_; ++c)
      wpack[n * 576 + tap * 64 + c] = f2bf(w[c * 9 + tap] * s);
}

// ---------------------------------------------------------------------------
// Kernel 1: implicit-GEMM 3x3 conv via v_wmma_f32_16x16x32_bf16.
// Block: 256 thr (8 waves), output tile 16 rows x 16 cols; wave w owns rows
// y0+2w, y0+2w+1 (two 16x16x32 M-tiles x 4 N-tiles = 8 accumulators).
// K = 576 in 18 steps of 32 (one conv tap per 2 steps).
// A tile in LDS [row][x][72 bf16-padded]  (46.7 KB, bank-conflict-free b128).
// B (72 KB packed weights) staged once via GLOBAL_LOAD_ASYNC_TO_LDS_B128 into
// dynamic LDS, padded to 584 bf16 per row (conflict-free, 16B-aligned).
// ---------------------------------------------------------------------------
__global__ __launch_bounds__(256) void conv_wmma(
    const float* __restrict__ feat,
    const unsigned short* __restrict__ wpack,
    const float* __restrict__ bias,
    float* __restrict__ conv52) {
  __shared__ __align__(16) unsigned short ldsA[18 * 18 * 72];  // 46656 B
  extern __shared__ u32x4 ldsBq[];                             // 64*584*2 = 74752 B
  const int tid  = threadIdx.x;
  const int lane = tid & 31, wv = tid >> 5;
  const int x0 = blockIdx.x << 4;
  const int y0 = blockIdx.y << 4;
  const int b  = blockIdx.z;

  // ---- stage packed weights -> LDS (async DMA path when available) ----
  char* ldsBc = (char*)ldsBq;
#ifdef HAVE_ASYNC_LDS
  for (int e = tid; e < 4608; e += 256) {            // 4608 x 16B chunks
    const int n = e / 72, r = e - n * 72;
    async_copy_b128((const char*)wpack + n * 1152 + r * 16,
                    ldsBc + n * 1168 + r * 16);
  }
#else
  for (int e = tid; e < 4608; e += 256) {
    const int n = e / 72, r = e - n * 72;
    *(u32x4*)(ldsBc + n * 1168 + r * 16) =
        *(const u32x4*)((const char*)wpack + n * 1152 + r * 16);
  }
#endif

  // ---- stage feature region rows [y0-1, y0+16], cols [x0-1, x0+16] ----
  for (int e = tid; e < 18 * 18 * 64; e += 256) {
    const int xr = e % 18;
    const int rc = e / 18;
    const int c  = rc & 63;
    const int row = rc >> 6;
    const int gy = y0 - 1 + row, gx = x0 - 1 + xr;
    float val = 0.f;
    if (gy >= 0 && gy < H_ && gx >= 0 && gx < W_)
      val = feat[((size_t)(b * C_ + c) * H_ + gy) * W_ + gx];
    ldsA[(row * 18 + xr) * 72 + c] = f2bf(val);
  }
#ifdef HAVE_ASYNC_LDS
  asm volatile("s_wait_asynccnt 0x0" ::: "memory");
#endif
  __syncthreads();

  const int half = lane >> 4, mrow = lane & 15;
  union AF { v16bf v; u32x4 q[2]; };
  v8f acc[2][4] = {};
  const u32x4* ldsA4 = (const u32x4*)ldsA;
  const u32x4* ldsB4 = (const u32x4*)ldsBq;

  for (int s = 0; s < 18; ++s) {
    const int tap = s >> 1, cb8 = (s & 1) << 2;      // cb in uint4 units (32 bf16 = 4)
    const int ki = tap / 3, kj = tap - ki * 3;
    // A 16x32 bf16 layout: half 0 -> K{0..7,16..23}, half 1 -> K{8..15,24..31}
    AF a[2];
#pragma unroll
    for (int t = 0; t < 2; ++t) {
      const int idx = ((wv * 2 + t + ki) * 18 + (mrow + kj)) * 9 + cb8;
      a[t].q[0] = ldsA4[idx + half];
      a[t].q[1] = ldsA4[idx + 2 + half];
    }
#pragma unroll
    for (int nt = 0; nt < 4; ++nt) {
      // B 32x16 layout: half 0 -> K{0..7,16..23}, half 1 -> K{8..15,24..31}, N = mrow.
      const int n  = (nt << 4) + mrow;
      const int bq = n * 73 + (s << 2) + half;  // 584 bf16 = 73 uint4 per n
      AF bfr;
      bfr.q[0] = ldsB4[bq];
      bfr.q[1] = ldsB4[bq + 2];
#pragma unroll
      for (int t = 0; t < 2; ++t)
        acc[t][nt] = __builtin_amdgcn_wmma_f32_16x16x32_bf16(
            false, a[t].v, false, bfr.v, (short)0, acc[t][nt], false, false);
    }
  }

  // D layout: lanes 0-15 N=lane M=r; lanes 16-31 N=lane-16 M=r+8.
#pragma unroll
  for (int t = 0; t < 2; ++t) {
    const int py = y0 + (wv << 1) + t;
#pragma unroll
    for (int nt = 0; nt < 4; ++nt) {
      const int ch = (nt << 4) + mrow;
      if (ch < 52) {
        const float bv = bias[ch];
#pragma unroll
        for (int r = 0; r < 8; ++r) {
          const int m = r + (half << 3);
          float v = acc[t][nt][r] + bv;
          if (ch >= 48) v = 1.f / (1.f + __expf(-v));   // sigmoid for att channels
          const size_t pix = ((size_t)b * H_ + py) * W_ + (x0 + m);
          conv52[pix * 52 + ch] = v;
        }
      }
    }
  }
}

// ---------------------------------------------------------------------------
// Kernel 2: per-pixel guide normalization + 49-tap propagation (iteration 5).
// Block (16,8) outputs; halo region 22x14; prod[49] per region pixel in LDS.
// ---------------------------------------------------------------------------
__global__ __launch_bounds__(128) void propagate(
    const float* __restrict__ conv52,
    const float* __restrict__ d0,
    const float* __restrict__ d00,
    float* __restrict__ out) {
  __shared__ float prodL[308 * 49];   // 60368 B
  __shared__ float g0L[308];          //  1232 B
  const int tx = threadIdx.x, ty = threadIdx.y;
  const int tid = ty * 16 + tx;
  const int x0 = blockIdx.x << 4;
  const int y0 = blockIdx.y << 3;
  const int b  = blockIdx.z;

  // SRC mapping: channel into conv52 (-1 => g1 pool), pool = 0:a7 1:a5 2:a3 3:a1
  constexpr int SRC_CH[49] = {
       0,  1,  2,  3,  4,  5,  6,
       7, 24, 25, 26, 27, 28,  8,
       9, 29, 40, 41, 42, 30, 10,
      11, 31, 43, -1, 44, 32, 12,
      13, 33, 45, 46, 47, 34, 14,
      15, 35, 36, 37, 38, 39, 16,
      17, 18, 19, 20, 21, 22, 23};
  constexpr int SRC_POOL[49] = {
      0,0,0,0,0,0,0,
      0,1,1,1,1,1,0,
      0,1,2,2,2,1,0,
      0,1,2,3,2,1,0,
      0,1,2,2,2,1,0,
      0,1,1,1,1,1,0,
      0,0,0,0,0,0,0};

  for (int p = tid; p < 308; p += 128) {
    const int rly = p / 22, rlx = p - rly * 22;
    const int gy = y0 - 3 + rly, gx = x0 - 3 + rlx;
    if (gy < 0 || gy >= H_ || gx < 0 || gx >= W_) {
#pragma unroll
      for (int t = 0; t < 49; ++t) prodL[p * 49 + t] = 0.f;
      g0L[p] = 0.f;
    } else {
      const size_t pix = ((size_t)b * H_ + gy) * W_ + gx;
      const float* c = conv52 + pix * 52;
      float att[4] = {c[48], c[49], c[50], c[51]};
      float s7a = 0.f, s7 = 0.f;
#pragma unroll
      for (int k = 0; k < 24; ++k) { const float v = c[k]; s7a += fabsf(v); s7 += v; }
      float s5a = 0.f, s5 = 0.f;
#pragma unroll
      for (int k = 24; k < 40; ++k) { const float v = c[k]; s5a += fabsf(v); s5 += v; }
      float s3a = 0.f, s3 = 0.f;
#pragma unroll
      for (int k = 40; k < 48; ++k) { const float v = c[k]; s3a += fabsf(v); s3 += v; }
      const float abssum = att[0]*s7a + att[1]*s5a + att[2]*s3a + att[3];
      const float gsum   = att[0]*s7  + att[1]*s5  + att[2]*s3  + att[3];
      const float inv = 1.f / abssum;      // abssum >= a1 > 0 (sigmoid outputs)
      const float base = inv * d0[pix];
      g0L[p] = 1.f - gsum * inv;
#pragma unroll
      for (int t = 0; t < 49; ++t) {
        const int ch = SRC_CH[t];
        const float val = (ch >= 0) ? c[ch] : 1.f;
        prodL[p * 49 + t] = att[SRC_POOL[t]] * val * base;
      }
    }
  }
  __syncthreads();

  // out(y,x) = g0*d00 + sum_t prod[t](y+3-i, x+3-j);  region row = ty+6-i
  const int py = y0 + ty, px = x0 + tx;
  const size_t pix = ((size_t)b * H_ + py) * W_ + px;
  float acc = g0L[(ty + 3) * 22 + (tx + 3)] * d00[pix];
#pragma unroll
  for (int i = 0; i < 7; ++i)
#pragma unroll
    for (int j = 0; j < 7; ++j)
      acc += prodL[((ty + 6 - i) * 22 + (tx + 6 - j)) * 49 + (i * 7 + j)];
  out[pix] = acc;
}

// ---------------------------------------------------------------------------
// Input order (setup_inputs insertion order, params flattened w,g,b,m,v per set):
//  0 feature | 1 d0 | 2 d00 | 3.. aff7, aff5, aff3, att7, att5, att3, att1
// ---------------------------------------------------------------------------
extern "C" void kernel_launch(void* const* d_in, const int* in_sizes, int n_in,
                              void* d_out, int out_size, void* d_ws, size_t ws_size,
                              hipStream_t stream) {
  (void)in_sizes; (void)n_in; (void)out_size; (void)ws_size;
  const float* feature = (const float*)d_in[0];
  const float* d0  = (const float*)d_in[1];
  const float* d00 = (const float*)d_in[2];
  PackArgs pa;
  for (int s = 0; s < 7; ++s) {
    const int base = 3 + s * 5;
    pa.w[s] = (const float*)d_in[base + 0];
    pa.g[s] = (const float*)d_in[base + 1];
    pa.b[s] = (const float*)d_in[base + 2];
    pa.m[s] = (const float*)d_in[base + 3];
    pa.v[s] = (const float*)d_in[base + 4];
  }
  const size_t npix = (size_t)B_ * H_ * W_;
  float*          conv52 = (float*)d_ws;                                   // 178 MB
  unsigned short* wpack  = (unsigned short*)((char*)d_ws + npix * 52 * sizeof(float));
  float*          bias   = (float*)((char*)wpack + 64 * 576 * sizeof(unsigned short));

  hipLaunchKernelGGL(pack_weights, dim3(1), dim3(64), 0, stream, pa, wpack, bias);
  hipLaunchKernelGGL(conv_wmma, dim3(W_ / 16, H_ / 16, B_), dim3(256), 64 * 584 * 2,
                     stream, feature, wpack, bias, conv52);
  hipLaunchKernelGGL(propagate, dim3(W_ / 16, H_ / 8, B_), dim3(16, 8, 1), 0, stream,
                     conv52, d0, d00, (float*)d_out);
}
